// TFSwinSelfAttention_55336358641801
// MI455X (gfx1250) — matrix operand order repaired and checked by reading, not compiled, not run on hardware
//
#include <hip/hip_runtime.h>
#include <hip/hip_bf16.h>
#include <math.h>

// ---------------------------------------------------------------------------
// Swin window self-attention for gfx1250 (MI455X), wave32 + v_wmma_f32_16x16x32_f16
//
// Shapes: B=4096 windows, N=49 tokens (padded to 64), D=128, H=4 heads, hd=32.
// Bandwidth-bound (~600MB traffic @ 23.3TB/s ~= 25us); all GEMM work in f16 WMMA.
// ---------------------------------------------------------------------------

typedef __attribute__((ext_vector_type(16))) _Float16 v16h;
typedef __attribute__((ext_vector_type(8)))  float    v8f;

#define WIN     7
#define N_TOK   49
#define N_PAD   64
#define NH      4
#define HD      32
#define D_MODEL 128
#define B_WIN   4096
#define NW      64

// ---------------------------------------------------------------------------
// Kernel 0: transpose + f16-convert the three weight matrices.
// WT[mat][j][i] = (f16) W[mat][i][j]  so that B-fragment loads are contiguous.
// ---------------------------------------------------------------------------
__global__ void convert_weights(const float* __restrict__ Wq,
                                const float* __restrict__ Wk,
                                const float* __restrict__ Wv,
                                _Float16* __restrict__ wt) {
  int idx = blockIdx.x * 256 + threadIdx.x;
  if (idx >= 3 * D_MODEL * D_MODEL) return;
  int mat = idx / (D_MODEL * D_MODEL);
  int r   = idx % (D_MODEL * D_MODEL);
  int i = r / D_MODEL, j = r % D_MODEL;
  const float* W = (mat == 0) ? Wq : ((mat == 1) ? Wk : Wv);
  wt[(size_t)mat * D_MODEL * D_MODEL + (size_t)j * D_MODEL + i] =
      (_Float16)W[(size_t)i * D_MODEL + j];
}

// ---------------------------------------------------------------------------
// Kernel 1: zero the pad regions (token rows 49..63 of q/k, cols 49..63 of vT)
// so padded-K WMMA contributions are exactly zero. One block per (b,h).
// ---------------------------------------------------------------------------
__global__ void zero_pads(_Float16* __restrict__ qw,
                          _Float16* __restrict__ kw,
                          _Float16* __restrict__ vtw) {
  const int bh = blockIdx.x;
  const int t  = threadIdx.x;
  const size_t base  = (size_t)bh * (N_PAD * HD);
  for (int i = t; i < (N_PAD - N_TOK) * HD; i += 128) {
    int row = N_TOK + i / HD, c = i % HD;
    qw[base + (size_t)row * HD + c] = (_Float16)0.f;
    kw[base + (size_t)row * HD + c] = (_Float16)0.f;
  }
  const size_t vbase = (size_t)bh * (HD * N_PAD);
  for (int i = t; i < HD * (N_PAD - N_TOK); i += 128) {
    int dd = i / (N_PAD - N_TOK), col = N_TOK + i % (N_PAD - N_TOK);
    vtw[vbase + (size_t)dd * N_PAD + col] = (_Float16)0.f;
  }
}

// ---------------------------------------------------------------------------
// Kernel 2: QKV projection. 200704x128 @ 128x128 (x3) with WMMA f16.
// Block = 256 threads = 8 waves; wave w owns output-column tile w (16 cols)
// and runs 4 K-steps per matrix (12 WMMAs/wave). Results scattered into
// head-split f16 q [b,h,64,32], k [b,h,64,32], vT [b,h,32,64].
// 200704 = 12544 * 16, so every m-tile is full.
// ---------------------------------------------------------------------------
__global__ __launch_bounds__(256) void qkv_proj(
    const float* __restrict__ hs,        // [200704, 128]
    const _Float16* __restrict__ wt,     // [3][128][128] (transposed, f16)
    const float* __restrict__ bq, const float* __restrict__ bk,
    const float* __restrict__ bv,
    _Float16* __restrict__ qw, _Float16* __restrict__ kw,
    _Float16* __restrict__ vtw) {
  const int lane   = threadIdx.x & 31;
  const int wvid   = threadIdx.x >> 5;   // n-tile 0..7
  const int l16    = lane & 15;
  const int half16 = lane >> 4;          // 0/1
  const int m0     = blockIdx.x * 16;

  // --- A fragments (shared across q/k/v): row m0+l16 of hidden, 4 K-steps ---
  const float* hrow = hs + (size_t)(m0 + l16) * D_MODEL;
  v16h a[4];
#pragma unroll
  for (int kt = 0; kt < 4; ++kt) {
    // lanes<16: K 0..7 & 16..23 ; lanes>=16: K 8..15 & 24..31 (A layout)
    const float4* c0 = (const float4*)(hrow + kt * 32 + half16 * 8);
    const float4* c1 = (const float4*)(hrow + kt * 32 + 16 + half16 * 8);
    float4 x0 = c0[0], x1 = c0[1], y0 = c1[0], y1 = c1[1];
    a[kt][0] = (_Float16)x0.x; a[kt][1] = (_Float16)x0.y;
    a[kt][2] = (_Float16)x0.z; a[kt][3] = (_Float16)x0.w;
    a[kt][4] = (_Float16)x1.x; a[kt][5] = (_Float16)x1.y;
    a[kt][6] = (_Float16)x1.z; a[kt][7] = (_Float16)x1.w;
    a[kt][8]  = (_Float16)y0.x; a[kt][9]  = (_Float16)y0.y;
    a[kt][10] = (_Float16)y0.z; a[kt][11] = (_Float16)y0.w;
    a[kt][12] = (_Float16)y1.x; a[kt][13] = (_Float16)y1.y;
    a[kt][14] = (_Float16)y1.z; a[kt][15] = (_Float16)y1.w;
  }

  const int col = wvid * 16 + l16;       // output column 0..127 (B-lane col)
  const int h   = col >> 5;
  const int dd  = col & 31;

  for (int mat = 0; mat < 3; ++mat) {
    const _Float16* wtm = wt + (size_t)mat * D_MODEL * D_MODEL
                             + (size_t)col * D_MODEL;
    v8f c = {};
#pragma unroll
    for (int kt = 0; kt < 4; ++kt) {
      // B layout: lane col, K 0..7 & 16..23 (lanes<16) / 8..15 & 24..31 (lanes>=16)
      const _Float16* src = wtm + kt * 32 + half16 * 8;
      v16h b;
#pragma unroll
      for (int e = 0; e < 16; ++e) b[e] = src[e < 8 ? e : e + 8];
      c = __builtin_amdgcn_wmma_f32_16x16x32_f16(false, a[kt], false, b,
                                                 (short)0, c, false, false);
    }
    const float* bias = (mat == 0) ? bq : ((mat == 1) ? bk : bv);
    const float  bb   = bias[col];
#pragma unroll
    for (int r = 0; r < 8; ++r) {
      const int g  = m0 + r + half16 * 8;     // global token row
      const int b_ = g / N_TOK;
      const int n  = g - b_ * N_TOK;
      const _Float16 hv = (_Float16)(c[r] + bb);
      const size_t bh = (size_t)(b_ * NH + h);
      if (mat == 0)      qw [bh * (N_PAD * HD) + (size_t)n * HD + dd] = hv;
      else if (mat == 1) kw [bh * (N_PAD * HD) + (size_t)n * HD + dd] = hv;
      else               vtw[bh * (HD * N_PAD) + (size_t)dd * N_PAD + n] = hv;
    }
  }
}

// ---------------------------------------------------------------------------
// Kernel 3: attention. One block (128 thr = 4 waves) per (b,h); wave w owns
// query tile rows [16w, 16w+16). scores = q*k^T (4 WMMA), softmax in C-frag
// registers (shfl_xor row reductions), P staged via 2KB LDS to re-shape into
// A-fragments, ctx = P*v (4 WMMA), f32 stores to out.
// ---------------------------------------------------------------------------
__global__ __launch_bounds__(128) void swin_attn(
    const _Float16* __restrict__ qw,    // [B*H][64][32]
    const _Float16* __restrict__ kw,    // [B*H][64][32]
    const _Float16* __restrict__ vtw,   // [B*H][32][64]
    const float* __restrict__ mask,     // [64][49][49]
    const float* __restrict__ relt,     // [169][4]
    float* __restrict__ out) {          // [4096][49][128]
  __shared__ _Float16 pbuf[4][16][N_PAD];   // 8 KB

  const int lane   = threadIdx.x & 31;
  const int wid    = threadIdx.x >> 5;      // m-tile 0..3
  const int l16    = lane & 15;
  const int half16 = lane >> 4;
  const int bh = blockIdx.x;
  const int b  = bh >> 2, h = bh & 3;
  const int m0 = wid * 16;

  const _Float16* qb = qw  + (size_t)bh * (N_PAD * HD);
  const _Float16* kb = kw  + (size_t)bh * (N_PAD * HD);
  const _Float16* vb = vtw + (size_t)bh * (HD * N_PAD);

  // ---- A fragment: q tile [16 x 32] ----
  v16h aq;
  {
    const _Float16* qr = qb + (size_t)(m0 + l16) * HD + half16 * 8;
#pragma unroll
    for (int e = 0; e < 8; ++e) aq[e] = qr[e];
#pragma unroll
    for (int e = 0; e < 8; ++e) aq[8 + e] = qr[16 + e];
  }

  // ---- scores = q * k^T : 4 column tiles of 16 keys ----
  v8f sc[4];
#pragma unroll
  for (int j = 0; j < 4; ++j) {
    const _Float16* kr = kb + (size_t)(j * 16 + l16) * HD + half16 * 8;
    v16h bk_;
#pragma unroll
    for (int e = 0; e < 16; ++e) bk_[e] = kr[e < 8 ? e : e + 8];
    v8f z = {};
    sc[j] = __builtin_amdgcn_wmma_f32_16x16x32_f16(false, aq, false, bk_,
                                                   (short)0, z, false, false);
  }

  // ---- scale + rel-pos bias + window mask + masked softmax (in C-frag) ----
  const float scale = 0.17677669529663687f;  // 1/sqrt(32)
  const float* mrow_base = mask + (size_t)(b & (NW - 1)) * N_TOK * N_TOK;
#pragma unroll
  for (int r = 0; r < 8; ++r) {
    const int m = m0 + r + half16 * 8;       // query row (padded 0..63)
    const int valid_m = (m < N_TOK);
    const int mi = m / WIN, mj = m % WIN;
    float mx = -3.0e38f;
#pragma unroll
    for (int j = 0; j < 4; ++j) {
      const int n = j * 16 + l16;            // key col (padded 0..63)
      float s;
      if (valid_m && n < N_TOK) {
        const int ni  = n / WIN, nj = n % WIN;
        const int idx = (mi - ni + WIN - 1) * (2 * WIN - 1) + (mj - nj + WIN - 1);
        s = sc[j][r] * scale + relt[idx * NH + h] + mrow_base[m * N_TOK + n];
      } else {
        s = -3.0e38f;
      }
      sc[j][r] = s;
      mx = fmaxf(mx, s);
    }
    // row lives across the 16-lane half-wave: xor masks 1,2,4,8 stay in-half
#pragma unroll
    for (int d = 1; d < 16; d <<= 1) mx = fmaxf(mx, __shfl_xor(mx, d));
    float sum = 0.f;
#pragma unroll
    for (int j = 0; j < 4; ++j) {
      const float e = __expf(sc[j][r] - mx);
      sc[j][r] = e;
      sum += e;
    }
#pragma unroll
    for (int d = 1; d < 16; d <<= 1) sum += __shfl_xor(sum, d);
    const float inv = 1.f / sum;
#pragma unroll
    for (int j = 0; j < 4; ++j) sc[j][r] *= inv;
  }

  // ---- stage P (f16) in LDS to re-shape D-frag -> A-frag ----
#pragma unroll
  for (int r = 0; r < 8; ++r) {
    const int row = r + half16 * 8;
#pragma unroll
    for (int j = 0; j < 4; ++j)
      pbuf[wid][row][j * 16 + l16] = (_Float16)sc[j][r];
  }
  __syncthreads();   // uniform; also orders LDS writes before reads

  // ---- ctx = P[16x64] * v[64x32] : 2 K-steps x 2 d-tiles ----
  v8f ctx0 = {}, ctx1 = {};
#pragma unroll
  for (int kt = 0; kt < 2; ++kt) {
    const _Float16* pr = &pbuf[wid][l16][kt * 32 + half16 * 8];
    v16h ap;
#pragma unroll
    for (int e = 0; e < 8; ++e) ap[e] = pr[e];
#pragma unroll
    for (int e = 0; e < 8; ++e) ap[8 + e] = pr[16 + e];

    const _Float16* vr0 = vb + (size_t)(0 * 16 + l16) * N_PAD + kt * 32 + half16 * 8;
    const _Float16* vr1 = vb + (size_t)(1 * 16 + l16) * N_PAD + kt * 32 + half16 * 8;
    v16h b0, b1;
#pragma unroll
    for (int e = 0; e < 16; ++e) { b0[e] = vr0[e < 8 ? e : e + 8]; b1[e] = vr1[e < 8 ? e : e + 8]; }
    ctx0 = __builtin_amdgcn_wmma_f32_16x16x32_f16(false, ap, false, b0,
                                                  (short)0, ctx0, false, false);
    ctx1 = __builtin_amdgcn_wmma_f32_16x16x32_f16(false, ap, false, b1,
                                                  (short)0, ctx1, false, false);
  }

  // ---- store ctx rows < 49 : out[b, n, h*32 + col] ----
  const int dcol0 = h * HD + l16;
  const int dcol1 = h * HD + 16 + l16;
#pragma unroll
  for (int r = 0; r < 8; ++r) {
    const int n = m0 + r + half16 * 8;
    if (n < N_TOK) {
      float* orow = out + ((size_t)b * N_TOK + n) * D_MODEL;
      orow[dcol0] = ctx0[r];
      orow[dcol1] = ctx1[r];
    }
  }
}

// ---------------------------------------------------------------------------
extern "C" void kernel_launch(void* const* d_in, const int* in_sizes, int n_in,
                              void* d_out, int out_size, void* d_ws, size_t ws_size,
                              hipStream_t stream) {
  const float* hs   = (const float*)d_in[0];   // [4096,49,128]
  const float* mask = (const float*)d_in[1];   // [64,49,49]
  const float* Wq   = (const float*)d_in[2];
  const float* bq   = (const float*)d_in[3];
  const float* Wk   = (const float*)d_in[4];
  const float* bk   = (const float*)d_in[5];
  const float* Wv   = (const float*)d_in[6];
  const float* bv   = (const float*)d_in[7];
  const float* relt = (const float*)d_in[8];   // [169,4]
  float* out = (float*)d_out;

  // Workspace layout (f16 elements):
  _Float16* ws = (_Float16*)d_ws;
  _Float16* wt  = ws;                                   // 3*128*128
  size_t off = (size_t)3 * D_MODEL * D_MODEL;
  _Float16* qw  = ws + off; off += (size_t)B_WIN * NH * N_PAD * HD;
  _Float16* kw  = ws + off; off += (size_t)B_WIN * NH * N_PAD * HD;
  _Float16* vtw = ws + off;                             // B*H*32*64

  convert_weights<<<(3 * D_MODEL * D_MODEL + 255) / 256, 256, 0, stream>>>(
      Wq, Wk, Wv, wt);
  zero_pads<<<B_WIN * NH, 128, 0, stream>>>(qw, kw, vtw);
  qkv_proj<<<(B_WIN * N_TOK) / 16, 256, 0, stream>>>(
      hs, wt, bq, bk, bv, qw, kw, vtw);
  swin_attn<<<B_WIN * NH, 128, 0, stream>>>(qw, kw, vtw, mask, relt, out);
}
